// NetVLAD_25048249270322
// MI455X (gfx1250) — matrix operand order repaired and checked by reading, not compiled, not run on hardware
//
#include <hip/hip_runtime.h>
#include <hip/hip_bf16.h>

typedef __attribute__((ext_vector_type(16))) __bf16        v16bf;
typedef __attribute__((ext_vector_type(8)))  float         v8f;
typedef __attribute__((ext_vector_type(8)))  unsigned int  v8u;

#define NV_N      64
#define NV_D      128
#define NV_P      4096
#define NV_K      32
#define NV_SPLITS 8
#define NV_PT     64
#define NV_TILES  ((NV_P / NV_SPLITS) / NV_PT)   // 8
#define NV_EPS    1e-12f

#if __has_builtin(__builtin_amdgcn_global_load_async_to_lds_b128)
#define NV_HAVE_ASYNC 1
#else
#define NV_HAVE_ASYNC 0
#endif

// Builtin takes (v4i* global_src, lds_dst, imm offset, imm cpol) on this toolchain.
typedef __attribute__((__vector_size__(16))) int nv_v4i;
typedef __attribute__((address_space(3))) nv_v4i* nv_ldsp;

static __device__ __forceinline__ void nv_wait_async0() {
#if __has_builtin(__builtin_amdgcn_s_wait_asynccnt)
  __builtin_amdgcn_s_wait_asynccnt(0);
#else
  asm volatile("s_wait_asynccnt 0x0" ::: "memory");
#endif
}

// float -> bf16, round-to-nearest-even
static __device__ __forceinline__ unsigned short f2bf(float f) {
  unsigned int u = __builtin_bit_cast(unsigned int, f);
  u += 0x7FFFu + ((u >> 16) & 1u);
  return (unsigned short)(u >> 16);
}

// ---- WMMA bf16 fragment loaders (ISA 7.12.2 layouts, wave32) ----
// A (16x32, MxK): lanes 0-15 -> M rows (K offset 0), lanes 16-31 -> same M, K offset +8.
// VGPR j<4 holds K = koff+2j(+1); j>=4 holds K = koff+16+2(j-4)(+1).
static __device__ __forceinline__ v16bf load_frag_a(const unsigned short* base,
                                                    int row_stride, int m0, int k0) {
  const int lane = threadIdx.x & 31;
  const unsigned short* row = base + (size_t)(m0 + (lane & 15)) * row_stride;
  const int koff = k0 + ((lane >> 4) << 3);
  v8u r;
#pragma unroll
  for (int j = 0; j < 4; ++j)
    r[j] = *(const unsigned int*)(row + koff + 2 * j);
#pragma unroll
  for (int j = 0; j < 4; ++j)
    r[4 + j] = *(const unsigned int*)(row + koff + 16 + 2 * j);
  return __builtin_bit_cast(v16bf, r);
}

// B (32x16, KxN): lane L -> column N=L%16; K order same as A (koff=8*(L/16), s<8: koff+s, s>=8: koff+16+s-8).
static __device__ __forceinline__ v16bf load_frag_b(const unsigned short* base,
                                                    int row_stride, int k0, int n0) {
  const int lane = threadIdx.x & 31;
  const unsigned short* col = base + n0 + (lane & 15);
  const int koff = k0 + ((lane >> 4) << 3);
  v8u r;
#pragma unroll
  for (int j = 0; j < 8; ++j) {
    const int k = koff + (j < 4 ? 2 * j : 16 + 2 * (j - 4));
    unsigned int lo = col[(size_t)k * row_stride];
    unsigned int hi = col[(size_t)(k + 1) * row_stride];
    r[j] = lo | (hi << 16);
  }
  return __builtin_bit_cast(v16bf, r);
}

// ---------------------------------------------------------------------------
// Stage 1: per (image, pixel-split) block. Streams pixels once from HBM via
// async global->LDS DMA (double buffered), produces partial agg/asum.
// ---------------------------------------------------------------------------
__global__ __launch_bounds__(256)
void netvlad_stage1(const float* __restrict__ x, const float* __restrict__ conv_w,
                    float* __restrict__ ws_agg, float* __restrict__ ws_asum) {
  __shared__ float          lds_xf[2][NV_D][NV_PT];     // raw f32 tiles     64 KB
  __shared__ unsigned short lds_xnb[NV_D][NV_PT];       // bf16, [d][p]      16 KB
  __shared__ unsigned short lds_xnt[NV_PT][NV_D];       // bf16, [p][d]      16 KB
  __shared__ unsigned short lds_wb[NV_K][NV_D];         // bf16 weights       8 KB
  __shared__ float          lds_a[NV_K][NV_PT];         // logits/softassign  8 KB
  __shared__ unsigned short lds_ab[NV_K][NV_PT];        // bf16 softassign    4 KB
  __shared__ float          lds_red[4][NV_PT];
  __shared__ float          lds_rn[NV_PT];

  const int tid  = threadIdx.x;
  const int lane = tid & 31;
  const int wave = tid >> 5;
  const int n    = blockIdx.y;
  const int s    = blockIdx.x;
  const int pbase0 = s * (NV_P / NV_SPLITS);
  const float* ximg = x + (size_t)n * NV_D * NV_P + pbase0;

  // Preload conv_w (32x128) as bf16 once.
  for (int i = tid; i < NV_K * NV_D; i += 256)
    (&lds_wb[0][0])[i] = f2bf(conv_w[i]);

  // Issue tile 0 into buffer 0 (async DMA global -> LDS, ASYNCcnt tracked).
#if NV_HAVE_ASYNC
  for (int i = tid; i < NV_D * (NV_PT / 4); i += 256) {
    int d = i >> 4, c4 = i & 15;
    __builtin_amdgcn_global_load_async_to_lds_b128(
        (nv_v4i*)(ximg + (size_t)d * NV_P + c4 * 4),
        (nv_ldsp)&lds_xf[0][d][c4 * 4], 0, 0);
  }
  nv_wait_async0();
#else
  for (int i = tid; i < NV_D * (NV_PT / 4); i += 256) {
    int d = i >> 4, c4 = i & 15;
    float4 v = *(const float4*)(ximg + (size_t)d * NV_P + c4 * 4);
    *(float4*)&lds_xf[0][d][c4 * 4] = v;
  }
#endif
  __syncthreads();

  const int mt  = wave & 1;        // cluster tile (0..1)
  const int nt  = wave >> 1;       // pixel tile for GEMM1 (0..3)
  const int dta = wave >> 1;       // this wave's agg d-tiles: dta and dta+4
  v8f accA = {}, accB = {};
  float asum_acc = 0.0f;

  for (int t = 0; t < NV_TILES; ++t) {
    const int buf = t & 1;

    // 0. Kick off async DMA for tile t+1 into the other buffer; it stays in
    //    flight underneath this whole iteration's LDS/WMMA work.
    if (t + 1 < NV_TILES) {
      const float* xnext = ximg + (t + 1) * NV_PT;
#if NV_HAVE_ASYNC
      for (int i = tid; i < NV_D * (NV_PT / 4); i += 256) {
        int d = i >> 4, c4 = i & 15;
        __builtin_amdgcn_global_load_async_to_lds_b128(
            (nv_v4i*)(xnext + (size_t)d * NV_P + c4 * 4),
            (nv_ldsp)&lds_xf[buf ^ 1][d][c4 * 4], 0, 0);
      }
#else
      for (int i = tid; i < NV_D * (NV_PT / 4); i += 256) {
        int d = i >> 4, c4 = i & 15;
        float4 v = *(const float4*)(xnext + (size_t)d * NV_P + c4 * 4);
        *(float4*)&lds_xf[buf ^ 1][d][c4 * 4] = v;
      }
#endif
      if (t + 2 < NV_TILES) {  // L2 prefetch two tiles ahead -> global_prefetch_b8
        const float* pf = ximg + (t + 2) * NV_PT + (size_t)(tid >> 1) * NV_P + (tid & 1) * 32;
        __builtin_prefetch(pf, 0, 3);
      }
    }

    // 1. Per-pixel L2 norms on the current buffer (4 partials per pixel).
    {
      int p = tid & 63, q = tid >> 6;
      float ss = 0.0f;
#pragma unroll 8
      for (int d = q * 32; d < q * 32 + 32; ++d) { float v = lds_xf[buf][d][p]; ss += v * v; }
      lds_red[q][p] = ss;
    }
    __syncthreads();
    if (tid < NV_PT) {
      float ss = lds_red[0][tid] + lds_red[1][tid] + lds_red[2][tid] + lds_red[3][tid];
      lds_rn[tid] = 1.0f / fmaxf(sqrtf(ss), NV_EPS);
    }
    __syncthreads();

    // 2. Normalize + convert to bf16 in both layouts.
    for (int i = tid; i < NV_D * NV_PT; i += 256) {
      int d = i >> 6, p = i & 63;
      unsigned short b = f2bf(lds_xf[buf][d][p] * lds_rn[p]);
      lds_xnb[d][p] = b;
      lds_xnt[p][d] = b;
    }
    __syncthreads();

    // 3. GEMM1: logits[32 x 64] = W[32x128] * Xn[128x64]; 8 waves = 2x4 tiles.
    {
      v8f acc = {};
#pragma unroll
      for (int c = 0; c < 4; ++c) {
        v16bf A = load_frag_a(&lds_wb[0][0], NV_D, mt * 16, c * 32);
        v16bf B = load_frag_b(&lds_xnb[0][0], NV_PT, c * 32, nt * 16);
        acc = __builtin_amdgcn_wmma_f32_16x16x32_bf16(false, A, false, B,
                                                      (short)0, acc, false, false);
      }
      const int m0 = mt * 16 + ((lane >> 4) << 3);
      const int nn = nt * 16 + (lane & 15);
#pragma unroll
      for (int r = 0; r < 8; ++r) lds_a[m0 + r][nn] = acc[r];
    }
    __syncthreads();

    // 4. Softmax over 32 clusters, one thread per pixel.
    if (tid < NV_PT) {
      const int p = tid;
      float mx = lds_a[0][p];
#pragma unroll
      for (int k = 1; k < NV_K; ++k) mx = fmaxf(mx, lds_a[k][p]);
      float sum = 0.0f;
#pragma unroll
      for (int k = 0; k < NV_K; ++k) sum += __expf(lds_a[k][p] - mx);
      const float inv = 1.0f / sum;
#pragma unroll
      for (int k = 0; k < NV_K; ++k) {
        float a = __expf(lds_a[k][p] - mx) * inv;
        lds_a[k][p]  = a;
        lds_ab[k][p] = f2bf(a);
      }
    }
    __syncthreads();

    // 5. asum accumulation (threads 0..31, fixed order -> deterministic).
    if (tid < NV_K) {
      float sacc = 0.0f;
#pragma unroll 8
      for (int p = 0; p < NV_PT; ++p) sacc += lds_a[tid][p];
      asum_acc += sacc;
    }

    // 6. GEMM2: agg[32x128] += A[32x64p] * Xn^T[64p x 128]; accumulate in regs.
#pragma unroll
    for (int pc = 0; pc < 2; ++pc) {
      v16bf A  = load_frag_a(&lds_ab[0][0], NV_PT, mt * 16, pc * 32);
      v16bf Ba = load_frag_b(&lds_xnt[0][0], NV_D, pc * 32, dta * 16);
      v16bf Bb = load_frag_b(&lds_xnt[0][0], NV_D, pc * 32, (dta + 4) * 16);
      accA = __builtin_amdgcn_wmma_f32_16x16x32_bf16(false, A, false, Ba,
                                                     (short)0, accA, false, false);
      accB = __builtin_amdgcn_wmma_f32_16x16x32_bf16(false, A, false, Bb,
                                                     (short)0, accB, false, false);
    }

    // 7. Drain this wave's async DMA for tile t+1, then block-wide barrier so
    //    the next iteration may read the fresh buffer.
#if NV_HAVE_ASYNC
    nv_wait_async0();
#endif
    __syncthreads();
  }

  // Write partial agg (C/D layout: lane 0-15 N=lane M=r; lane 16-31 N=lane-16 M=r+8).
  float* aggp = ws_agg + (size_t)(n * NV_SPLITS + s) * NV_K * NV_D;
  const int m0 = mt * 16 + ((lane >> 4) << 3);
  const int nn = lane & 15;
#pragma unroll
  for (int r = 0; r < 8; ++r) {
    aggp[(m0 + r) * NV_D + dta * 16 + nn]       = accA[r];
    aggp[(m0 + r) * NV_D + (dta + 4) * 16 + nn] = accB[r];
  }
  if (tid < NV_K) ws_asum[(n * NV_SPLITS + s) * NV_K + tid] = asum_acc;
}

// ---------------------------------------------------------------------------
// Stage 2: reduce splits, vlad = agg - asum*centroids, intra-norm, global norm.
// ---------------------------------------------------------------------------
__global__ __launch_bounds__(256)
void netvlad_finalize(const float* __restrict__ ws_agg, const float* __restrict__ ws_asum,
                      const float* __restrict__ centroids, float* __restrict__ out) {
  __shared__ float asum[NV_K];
  __shared__ float red[NV_K][8];
  __shared__ float rk[NV_K];
  __shared__ float gred[256];
  __shared__ float rgs;

  const int n   = blockIdx.x;
  const int tid = threadIdx.x;
  const int k   = tid >> 3;     // 8 threads per cluster
  const int sub = tid & 7;      // each owns 16 consecutive d

  if (tid < NV_K) {
    float sa = 0.0f;
#pragma unroll
    for (int sp = 0; sp < NV_SPLITS; ++sp) sa += ws_asum[(n * NV_SPLITS + sp) * NV_K + tid];
    asum[tid] = sa;
  }
  __syncthreads();

  const float* aggn = ws_agg + (size_t)n * NV_SPLITS * NV_K * NV_D;
  float vals[16];
  float ssq = 0.0f;
#pragma unroll
  for (int i = 0; i < 16; ++i) {
    const int idx = k * NV_D + sub * 16 + i;
    float a = 0.0f;
#pragma unroll
    for (int sp = 0; sp < NV_SPLITS; ++sp) a += aggn[sp * NV_K * NV_D + idx];
    float v = a - asum[k] * centroids[idx];
    vals[i] = v;
    ssq += v * v;
  }
  red[k][sub] = ssq;
  __syncthreads();
  if (tid < NV_K) {
    float s = 0.0f;
#pragma unroll
    for (int j = 0; j < 8; ++j) s += red[tid][j];
    rk[tid] = 1.0f / fmaxf(sqrtf(s), NV_EPS);
  }
  __syncthreads();

  const float r = rk[k];
  float g = 0.0f;
#pragma unroll
  for (int i = 0; i < 16; ++i) { vals[i] *= r; g += vals[i] * vals[i]; }
  gred[tid] = g;
  __syncthreads();
  for (int s2 = 128; s2 > 0; s2 >>= 1) {
    if (tid < s2) gred[tid] += gred[tid + s2];
    __syncthreads();
  }
  if (tid == 0) rgs = 1.0f / fmaxf(sqrtf(gred[0]), NV_EPS);
  __syncthreads();
  const float rg = rgs;
#pragma unroll
  for (int i = 0; i < 16; ++i)
    out[(size_t)n * (NV_K * NV_D) + k * NV_D + sub * 16 + i] = vals[i] * rg;
}

// ---------------------------------------------------------------------------
extern "C" void kernel_launch(void* const* d_in, const int* in_sizes, int n_in,
                              void* d_out, int out_size, void* d_ws, size_t ws_size,
                              hipStream_t stream) {
  (void)in_sizes; (void)n_in; (void)out_size; (void)ws_size;
  const float* x         = (const float*)d_in[0];
  const float* conv_w    = (const float*)d_in[1];
  const float* centroids = (const float*)d_in[2];
  float* out = (float*)d_out;

  float* ws_agg  = (float*)d_ws;                                    // 64*8*32*128 f32 = 8 MB
  float* ws_asum = ws_agg + (size_t)NV_N * NV_SPLITS * NV_K * NV_D; // 64*8*32 f32

  netvlad_stage1<<<dim3(NV_SPLITS, NV_N), 256, 0, stream>>>(x, conv_w, ws_agg, ws_asum);
  netvlad_finalize<<<NV_N, 256, 0, stream>>>(ws_agg, ws_asum, centroids, out);
}
